// TransformerLayer_3753801417325
// MI455X (gfx1250) — matrix operand order repaired and checked by reading, not compiled, not run on hardware
//
#include <hip/hip_runtime.h>
#include <hip/hip_bf16.h>

// ---------------------------------------------------------------------------
// Transformer layer for MI455X (gfx1250), wave32 + WMMA bf16 + async/TDM.
//   B=2 S=2048 D=2048 Dff=8192, H=16 KVH=8 HD=128, fp32 in/out.
// Memory-bound (~0.4GB @23.3TB/s): bf16 WMMA GEMMs with on-the-fly fp32->bf16
// weight conversion, GEMM A-tiles via global_load_async_to_lds_b128 (ASYNCcnt),
// flash-attention Q/V tiles via TDM tensor_load_to_lds (TENSORcnt),
// fused dual gate/up GEMM with SiLU epilogue, fused residual epilogues.
// ---------------------------------------------------------------------------

typedef __attribute__((ext_vector_type(16))) __bf16 v16bf;
typedef __attribute__((ext_vector_type(8)))  __bf16 v8bf;
typedef __attribute__((ext_vector_type(8)))  float  v8f;
typedef __attribute__((ext_vector_type(4)))  unsigned int u32x4;
typedef __attribute__((ext_vector_type(8)))  unsigned int u32x8;

#define DMODEL 2048
#define DFF    8192
#define NH     16
#define NKVH   8
#define HD     128
#define SEQ    2048
#define BATCH  2
#define MROWS  (BATCH*SEQ)   // 4096

static __device__ __forceinline__ v8f zero_v8f() {
  v8f z;
#pragma unroll
  for (int i = 0; i < 8; ++i) z[i] = 0.0f;
  return z;
}

static __device__ __forceinline__ v8f wmma_bf16(v16bf a, v16bf b, v8f c) {
  // D = A(16x32 bf16) * B(32x16 bf16) + C(16x16 f32)
  return __builtin_amdgcn_wmma_f32_16x16x32_bf16(
      /*neg_a=*/false, a, /*neg_b=*/false, b,
      /*c_mod=*/(short)0, c, /*reuse_a=*/false, /*reuse_b=*/false);
}

// A-matrix fragment (16x32), LDS row-major tile, row stride `ld` elements.
// lane<16: row=lane, K 0..7 & 16..23 ; lane>=16: row=lane-16, K 8..15 & 24..31
static __device__ __forceinline__ v16bf frag_a(const __bf16* p, int ld) {
  int lane = threadIdx.x & 31;
  const __bf16* q = p + (lane & 15) * ld + ((lane >> 4) << 3);
  v8bf lo = *(const v8bf*)q;
  v8bf hi = *(const v8bf*)(q + 16);
  v16bf r;
#pragma unroll
  for (int i = 0; i < 8; ++i) { r[i] = lo[i]; r[i + 8] = hi[i]; }
  return r;
}

// B-matrix fragment (32x16), LDS row-major [K][N] tile; p points at column 0
// of the 16-wide subtile. lane holds column n=lane&15, K 0..7 & 16..23 (lane<16) or 8..15 & 24..31.
static __device__ __forceinline__ v16bf frag_b(const __bf16* p, int ld) {
  int lane = threadIdx.x & 31;
  const __bf16* q = p + (lane & 15) + ((lane >> 4) << 3) * ld;
  v16bf r;
#pragma unroll
  for (int i = 0; i < 8; ++i) { r[i] = q[i * ld]; r[i + 8] = q[(i + 16) * ld]; }
  return r;
}

// ---------------------------------------------------------------------------
// TDM: 2D tile load global->LDS (bf16 elements, data_size=2B).
// D# group0 (4 SGPRs) + group1 (8 SGPRs); VADDR2/VADDR3 = NULL (<=2D tensor).
// Wave-level DMA, EXEC ignored: call from ONE wave; complete with TENSORcnt.
// ---------------------------------------------------------------------------
static __device__ __forceinline__ void tdm_load_2d_bf16(
    unsigned lds_off, const void* gptr, unsigned tile_w, unsigned tile_h,
    unsigned row_stride_elems) {
  unsigned long long ga = (unsigned long long)(uintptr_t)gptr;
  const unsigned td0 = 0x7FFFFFFFu, td1 = 0x7FFFFFFFu;  // huge dims: no OOB clip
  u32x4 g0;
  g0[0] = 1u;                                   // count=1, user descriptor
  g0[1] = lds_off;                              // lds_addr (bytes)
  g0[2] = (unsigned)ga;                         // global_addr[31:0]
  g0[3] = (unsigned)((ga >> 32) & 0x01FFFFFFu)  // global_addr[56:32]
          | (2u << 30);                         // type = 2 ("image")
  u32x8 g1;
  g1[0] = (1u << 16);                           // data_size=1 -> 2 bytes/elem
  g1[1] = (td0 & 0xFFFFu) << 16;                // tensor_dim0[15:0]
  g1[2] = (td0 >> 16) | ((td1 & 0xFFFFu) << 16);// tensor_dim0[31:16]|dim1[15:0]
  g1[3] = (td1 >> 16) | (tile_w << 16);         // tensor_dim1[31:16]|tile_dim0
  g1[4] = tile_h & 0xFFFFu;                     // tile_dim1 (tile_dim2=0)
  g1[5] = row_stride_elems;                     // tensor_dim0_stride[31:0]
  g1[6] = 0u;                                   // stride[47:32]|dim1_stride lo
  g1[7] = 0u;
  asm volatile("tensor_load_to_lds %0, %1" :: "s"(g0), "s"(g1) : "memory");
}

// Per-lane async 16B copy global->LDS (ASYNCcnt). INST_OFFSET applies to both.
static __device__ __forceinline__ void async_copy_b128(
    unsigned lds_off, const void* gptr) {
  unsigned long long ga = (unsigned long long)(uintptr_t)gptr;
  asm volatile("global_load_async_to_lds_b128 %0, %1, off"
               :: "v"(lds_off), "v"(ga) : "memory");
}

// ---------------------------------------------------------------------------
// RMSNorm: one block per row of D=2048, output bf16.
// ---------------------------------------------------------------------------
__global__ __launch_bounds__(256) void rmsnorm_kernel(
    const float* __restrict__ X, const float* __restrict__ W,
    __bf16* __restrict__ Out) {
  __shared__ float red[8];
  size_t row = blockIdx.x;
  const float* x = X + row * DMODEL;
  int base = threadIdx.x * 8;
  float v[8];
  float ss = 0.0f;
#pragma unroll
  for (int i = 0; i < 8; ++i) { v[i] = x[base + i]; ss += v[i] * v[i]; }
#pragma unroll
  for (int d = 16; d >= 1; d >>= 1) ss += __shfl_xor(ss, d, 32);
  if ((threadIdx.x & 31) == 0) red[threadIdx.x >> 5] = ss;
  __syncthreads();
  float tot = 0.0f;
#pragma unroll
  for (int i = 0; i < 8; ++i) tot += red[i];
  float rstd = rsqrtf(tot * (1.0f / DMODEL) + 1e-5f);
  __bf16* o = Out + row * DMODEL;
#pragma unroll
  for (int i = 0; i < 8; ++i) o[base + i] = (__bf16)(v[i] * rstd * W[base + i]);
}

// ---------------------------------------------------------------------------
// GEMM: C(MxN) = A(MxK bf16) * B(KxN f32->bf16).   128x128 block tile, BK=32,
// 256 threads = 8 waves (4x2), each wave: 32x64 via 8 (or 16 dual) WMMA/kstep.
// A tile: global_load_async_to_lds_b128; B tile: VGPR-staged fp32->bf16.
// MODE 0: store f32   MODE 1: f32 + residual   MODE 2: dual-B, SiLU(g)*u -> bf16
// ---------------------------------------------------------------------------
template <int MODE>
__global__ __launch_bounds__(256) void gemm_kernel(
    const __bf16* __restrict__ A, const float* __restrict__ B0,
    const float* __restrict__ B1, const float* __restrict__ resid,
    float* __restrict__ outf, __bf16* __restrict__ outb,
    int N, int K) {
  __shared__ __align__(16) __bf16 As[128 * 32];
  __shared__ __align__(16) __bf16 Bs[32 * 128];
  __shared__ __align__(16) __bf16 B2s[32 * 128];

  const int m0 = blockIdx.y * 128, n0 = blockIdx.x * 128;
  const int tid = threadIdx.x, lane = tid & 31, wid = tid >> 5;
  const int wm = (wid & 3) * 32, wn = (wid >> 2) * 64;

  v8f acc[2][4], acc2[2][4];
#pragma unroll
  for (int i = 0; i < 2; ++i)
#pragma unroll
    for (int j = 0; j < 4; ++j) { acc[i][j] = zero_v8f(); acc2[i][j] = zero_v8f(); }

  const int ar = tid >> 1, ac = (tid & 1) * 16;   // A: 128 rows x 32 cols
  const int br = tid >> 3, bc = (tid & 7) * 16;   // B: 32 rows x 128 cols
  const unsigned lds_a = (unsigned)(uintptr_t)(As + ar * 32 + ac);

  for (int k0 = 0; k0 < K; k0 += 32) {
    __syncthreads();
    {  // A tile (bf16): async copy to LDS, overlapped with B staging below
      const __bf16* src = A + (size_t)(m0 + ar) * K + k0 + ac;
      async_copy_b128(lds_a, src);
      async_copy_b128(lds_a + 16, src + 8);
    }
    {  // B0 tile: fp32 -> bf16 through VGPRs
      const float* src = B0 + (size_t)(k0 + br) * N + n0 + bc;
      float4 f0 = ((const float4*)src)[0], f1 = ((const float4*)src)[1];
      float4 f2 = ((const float4*)src)[2], f3 = ((const float4*)src)[3];
      v8bf t0, t1;
      t0[0]=(__bf16)f0.x; t0[1]=(__bf16)f0.y; t0[2]=(__bf16)f0.z; t0[3]=(__bf16)f0.w;
      t0[4]=(__bf16)f1.x; t0[5]=(__bf16)f1.y; t0[6]=(__bf16)f1.z; t0[7]=(__bf16)f1.w;
      t1[0]=(__bf16)f2.x; t1[1]=(__bf16)f2.y; t1[2]=(__bf16)f2.z; t1[3]=(__bf16)f2.w;
      t1[4]=(__bf16)f3.x; t1[5]=(__bf16)f3.y; t1[6]=(__bf16)f3.z; t1[7]=(__bf16)f3.w;
      *(v8bf*)(Bs + br * 128 + bc)     = t0;
      *(v8bf*)(Bs + br * 128 + bc + 8) = t1;
    }
    if (MODE == 2) {  // B1 tile
      const float* src = B1 + (size_t)(k0 + br) * N + n0 + bc;
      float4 f0 = ((const float4*)src)[0], f1 = ((const float4*)src)[1];
      float4 f2 = ((const float4*)src)[2], f3 = ((const float4*)src)[3];
      v8bf t0, t1;
      t0[0]=(__bf16)f0.x; t0[1]=(__bf16)f0.y; t0[2]=(__bf16)f0.z; t0[3]=(__bf16)f0.w;
      t0[4]=(__bf16)f1.x; t0[5]=(__bf16)f1.y; t0[6]=(__bf16)f1.z; t0[7]=(__bf16)f1.w;
      t1[0]=(__bf16)f2.x; t1[1]=(__bf16)f2.y; t1[2]=(__bf16)f2.z; t1[3]=(__bf16)f2.w;
      t1[4]=(__bf16)f3.x; t1[5]=(__bf16)f3.y; t1[6]=(__bf16)f3.z; t1[7]=(__bf16)f3.w;
      *(v8bf*)(B2s + br * 128 + bc)     = t0;
      *(v8bf*)(B2s + br * 128 + bc + 8) = t1;
    }
    if (k0 + 32 < K)  // global_prefetch_b8 for next weight tile
      __builtin_prefetch(B0 + (size_t)(k0 + 32 + br) * N + n0 + bc, 0, 1);
    asm volatile("s_wait_asynccnt 0" ::: "memory");  // A tile landed in LDS
    __syncthreads();

    v16bf a0 = frag_a(As + wm * 32, 32);
    v16bf a1 = frag_a(As + (wm + 16) * 32, 32);
#pragma unroll
    for (int nt = 0; nt < 4; ++nt) {
      v16bf b = frag_b(Bs + wn + nt * 16, 128);
      acc[0][nt] = wmma_bf16(a0, b, acc[0][nt]);
      acc[1][nt] = wmma_bf16(a1, b, acc[1][nt]);
      if (MODE == 2) {
        v16bf b2 = frag_b(B2s + wn + nt * 16, 128);
        acc2[0][nt] = wmma_bf16(a0, b2, acc2[0][nt]);
        acc2[1][nt] = wmma_bf16(a1, b2, acc2[1][nt]);
      }
    }
  }

  // Epilogue. C layout: VGPR j -> row j (+8 for lanes>=16), col = lane&15.
  const int rup = ((lane >> 4) << 3), cl = lane & 15;
#pragma unroll
  for (int sm = 0; sm < 2; ++sm)
#pragma unroll
    for (int nt = 0; nt < 4; ++nt)
#pragma unroll
      for (int j = 0; j < 8; ++j) {
        int grow = m0 + wm + sm * 16 + j + rup;
        int gcol = n0 + wn + nt * 16 + cl;
        size_t idx = (size_t)grow * N + gcol;
        float c = acc[sm][nt][j];
        if (MODE == 0) {
          outf[idx] = c;
        } else if (MODE == 1) {
          outf[idx] = c + resid[idx];
        } else {
          float g = c, u = acc2[sm][nt][j];
          float s = g / (1.0f + __expf(-g));   // SiLU
          outb[idx] = (__bf16)(s * u);
        }
      }
}

// ---------------------------------------------------------------------------
// RoPE (interleaved-halves form of the reference) + f32->bf16.
// ---------------------------------------------------------------------------
__global__ __launch_bounds__(64) void rope_kernel(
    const float* __restrict__ X, const float* __restrict__ C,
    const float* __restrict__ Sn, __bf16* __restrict__ Out, int nheads) {
  int d = threadIdx.x;                       // 0..63
  int h = blockIdx.x, s = blockIdx.y, b = blockIdx.z;
  size_t row = (size_t)b * SEQ + s;
  const float* x = X + row * (size_t)(nheads * HD) + h * HD;
  __bf16* o = Out + row * (size_t)(nheads * HD) + h * HD;
  float c = C[row * 64 + d], sn = Sn[row * 64 + d];
  float x1 = x[2 * d], x2 = x[2 * d + 1];
  o[d]      = (__bf16)(x1 * c - x2 * sn);
  o[64 + d] = (__bf16)(x1 * sn + x2 * c);
}

__global__ __launch_bounds__(256) void f32_to_bf16_kernel(
    const float* __restrict__ in, __bf16* __restrict__ out, size_t n) {
  size_t i = ((size_t)blockIdx.x * blockDim.x + threadIdx.x) * 4;
  if (i + 3 < n) {
    float4 f = *(const float4*)(in + i);
    out[i]     = (__bf16)f.x; out[i + 1] = (__bf16)f.y;
    out[i + 2] = (__bf16)f.z; out[i + 3] = (__bf16)f.w;
  }
}

// ---------------------------------------------------------------------------
// Flash attention (GQA n_rep=2). grid(S/64, NH, B), 128 threads = 4 waves.
// Q and V tiles DMA'd by the TDM (wave 0); K manually transposed into LDS.
// QK^T and PV via WMMA bf16; streaming softmax with half-wave shuffles.
// ---------------------------------------------------------------------------
__global__ __launch_bounds__(128) void flash_attn_kernel(
    const __bf16* __restrict__ Q, const __bf16* __restrict__ Kb,
    const __bf16* __restrict__ Vb, const float* __restrict__ mask,
    __bf16* __restrict__ O) {
  __shared__ __align__(16) __bf16 Qs[64 * 128];    // [qrow][d]
  __shared__ __align__(16) __bf16 Kts[128 * 64];   // [d][kv]  (transposed)
  __shared__ __align__(16) __bf16 Vs[64 * 128];    // [kv][d]
  __shared__ __align__(16) __bf16 Ps[4 * 16 * 64]; // per-wave P tile

  const int q0 = blockIdx.x * 64, h = blockIdx.y, b = blockIdx.z;
  const int kvh = h >> 1;   // n_rep = 2
  const int tid = threadIdx.x, lane = tid & 31, wid = tid >> 5;
  const int r0 = wid * 16;
  const int rup = ((lane >> 4) << 3), cl = lane & 15;
  __bf16* Pw = Ps + wid * 16 * 64;
  const float scale = 0.08838834764831845f;  // 128^-0.5

  if (wid == 0) {  // Q tile once, via TDM
    tdm_load_2d_bf16((unsigned)(uintptr_t)Qs,
                     Q + ((size_t)(b * SEQ + q0)) * DMODEL + h * HD,
                     /*tile_w=*/HD, /*tile_h=*/64, /*stride=*/DMODEL);
  }

  float m_i[8], l_i[8];
  v8f o_acc[8];
#pragma unroll
  for (int j = 0; j < 8; ++j) { m_i[j] = -1e30f; l_i[j] = 0.0f; }
#pragma unroll
  for (int t = 0; t < 8; ++t) o_acc[t] = zero_v8f();

  for (int j0 = 0; j0 < SEQ; j0 += 64) {
    __syncthreads();  // previous iteration's LDS reads complete
    if (wid == 0) {   // V tile via TDM, in flight during K transpose below
      tdm_load_2d_bf16((unsigned)(uintptr_t)Vs,
                       Vb + ((size_t)(b * SEQ + j0)) * (NKVH * HD) + kvh * HD,
                       /*tile_w=*/HD, /*tile_h=*/64, /*stride=*/NKVH * HD);
    }
    {  // K tile transposed into LDS (all 4 waves)
      int r = tid >> 1, c0 = (tid & 1) * 64;
      const __bf16* ks =
          Kb + ((size_t)(b * SEQ + j0 + r)) * (NKVH * HD) + kvh * HD + c0;
#pragma unroll
      for (int i = 0; i < 8; ++i) {
        v8bf kv = *(const v8bf*)(ks + i * 8);
#pragma unroll
        for (int e = 0; e < 8; ++e) Kts[(c0 + i * 8 + e) * 64 + r] = kv[e];
      }
    }
    if (wid == 0) __builtin_amdgcn_s_wait_tensorcnt(0);  // Q (first iter) + V
    __syncthreads();

    // S = Q * K^T  (wave: 16 q-rows x 64 kv-cols)
    v8f sacc[4];
#pragma unroll
    for (int nt = 0; nt < 4; ++nt) sacc[nt] = zero_v8f();
#pragma unroll
    for (int ks = 0; ks < 4; ++ks) {
      v16bf aq = frag_a(Qs + r0 * 128 + ks * 32, 128);
#pragma unroll
      for (int nt = 0; nt < 4; ++nt) {
        v16bf bk = frag_b(Kts + (ks * 32) * 64 + nt * 16, 64);
        sacc[nt] = wmma_bf16(aq, bk, sacc[nt]);
      }
    }

    // streaming softmax; each lane owns 8 rows (j + rup) over cols nt*16+cl
#pragma unroll
    for (int j = 0; j < 8; ++j) {
      int grow = q0 + r0 + j + rup;
      const float* mrow = mask + ((size_t)(b * SEQ + grow)) * SEQ + j0;
      float sv[4];
      float mx = -1e30f;
#pragma unroll
      for (int nt = 0; nt < 4; ++nt) {
        sv[nt] = sacc[nt][j] * scale + mrow[nt * 16 + cl];
        mx = fmaxf(mx, sv[nt]);
      }
#pragma unroll
      for (int d = 8; d >= 1; d >>= 1) mx = fmaxf(mx, __shfl_xor(mx, d, 32));
      float mnew = fmaxf(m_i[j], mx);
      float f = __expf(m_i[j] - mnew);
      float rs = 0.0f;
      float pe[4];
#pragma unroll
      for (int nt = 0; nt < 4; ++nt) { pe[nt] = __expf(sv[nt] - mnew); rs += pe[nt]; }
#pragma unroll
      for (int d = 8; d >= 1; d >>= 1) rs += __shfl_xor(rs, d, 32);
      l_i[j] = l_i[j] * f + rs;
      m_i[j] = mnew;
#pragma unroll
      for (int t = 0; t < 8; ++t) o_acc[t][j] *= f;
#pragma unroll
      for (int nt = 0; nt < 4; ++nt)
        Pw[(j + rup) * 64 + nt * 16 + cl] = (__bf16)pe[nt];
    }
    asm volatile("s_wait_dscnt 0" ::: "memory");  // P visible to own wave

    // O += P * V
#pragma unroll
    for (int ks = 0; ks < 2; ++ks) {
      v16bf ap = frag_a(Pw + ks * 32, 64);
#pragma unroll
      for (int nt = 0; nt < 8; ++nt) {
        v16bf bv = frag_b(Vs + (ks * 32) * 128 + nt * 16, 128);
        o_acc[nt] = wmma_bf16(ap, bv, o_acc[nt]);
      }
    }
  }

  // epilogue: O /= l, write bf16 at (b, s, h*128 + d)
#pragma unroll
  for (int nt = 0; nt < 8; ++nt)
#pragma unroll
    for (int j = 0; j < 8; ++j) {
      int grow = q0 + r0 + j + rup;
      float val = o_acc[nt][j] / l_i[j];
      O[((size_t)(b * SEQ + grow)) * DMODEL + h * HD + nt * 16 + cl] = (__bf16)val;
    }
}

// ---------------------------------------------------------------------------
// Launch
// ---------------------------------------------------------------------------
extern "C" void kernel_launch(void* const* d_in, const int* in_sizes, int n_in,
                              void* d_out, int out_size, void* d_ws, size_t ws_size,
                              hipStream_t stream) {
  const float* hidden = (const float*)d_in[0];
  const float* cosb   = (const float*)d_in[1];
  const float* sinb   = (const float*)d_in[2];
  const float* amask  = (const float*)d_in[3];
  const float* wq     = (const float*)d_in[4];
  const float* wk     = (const float*)d_in[5];
  const float* wv     = (const float*)d_in[6];
  const float* wo     = (const float*)d_in[7];
  const float* wgate  = (const float*)d_in[8];
  const float* wup    = (const float*)d_in[9];
  const float* wdown  = (const float*)d_in[10];
  const float* ln1w   = (const float*)d_in[11];
  const float* ln2w   = (const float*)d_in[12];
  float* out = (float*)d_out;

  const size_t M = MROWS;
  char* W = (char*)d_ws;
  size_t off = 0;
  auto take = [&](size_t bytes) {
    size_t o = off;
    off += (bytes + 255) & ~(size_t)255;
    return o;
  };
  __bf16* xn  = (__bf16*)(W + take(M * DMODEL * 2));
  float*  qf  = (float*)(W + take(M * DMODEL * 4));
  float*  kf  = (float*)(W + take(M * (NKVH * HD) * 4));
  float*  vf  = (float*)(W + take(M * (NKVH * HD) * 4));
  __bf16* qb  = (__bf16*)(W + take(M * DMODEL * 2));
  __bf16* kb  = (__bf16*)(W + take(M * (NKVH * HD) * 2));
  __bf16* vb  = (__bf16*)(W + take(M * (NKVH * HD) * 2));
  __bf16* xn2 = (__bf16*)(W + take(M * DMODEL * 2));
  __bf16* hgl = (__bf16*)(W + take(M * DFF * 2));
  // reuse regions freed mid-pipeline:
  __bf16* attnb = (__bf16*)qf;   // qf dead after RoPE(q)
  float*  hs2   = kf;            // kf+vf contiguous 32MB, dead after RoPE/convert

  dim3 blk256(256), blk128(128), blk64(64);

  // 1) x = rms_norm(hidden, ln1) -> bf16
  rmsnorm_kernel<<<M, blk256, 0, stream>>>(hidden, ln1w, xn);

  // 2) q/k/v projections (f32 out, RoPE still pending)
  gemm_kernel<0><<<dim3(DMODEL / 128, M / 128), blk256, 0, stream>>>(
      xn, wq, nullptr, nullptr, qf, nullptr, DMODEL, DMODEL);
  gemm_kernel<0><<<dim3((NKVH * HD) / 128, M / 128), blk256, 0, stream>>>(
      xn, wk, nullptr, nullptr, kf, nullptr, NKVH * HD, DMODEL);
  gemm_kernel<0><<<dim3((NKVH * HD) / 128, M / 128), blk256, 0, stream>>>(
      xn, wv, nullptr, nullptr, vf, nullptr, NKVH * HD, DMODEL);

  // 3) RoPE q,k (-> bf16); convert v -> bf16
  rope_kernel<<<dim3(NH, SEQ, BATCH), blk64, 0, stream>>>(qf, cosb, sinb, qb, NH);
  rope_kernel<<<dim3(NKVH, SEQ, BATCH), blk64, 0, stream>>>(kf, cosb, sinb, kb, NKVH);
  f32_to_bf16_kernel<<<(M * NKVH * HD) / (256 * 4), blk256, 0, stream>>>(
      vf, vb, M * NKVH * HD);

  // 4) flash attention -> attnb (bf16, (B,S,H*HD))
  flash_attn_kernel<<<dim3(SEQ / 64, NH, BATCH), blk128, 0, stream>>>(
      qb, kb, vb, amask, attnb);

  // 5) hs2 = hidden + attn @ wo
  gemm_kernel<1><<<dim3(DMODEL / 128, M / 128), blk256, 0, stream>>>(
      attnb, wo, nullptr, hidden, hs2, nullptr, DMODEL, DMODEL);

  // 6) x2 = rms_norm(hs2, ln2) -> bf16
  rmsnorm_kernel<<<M, blk256, 0, stream>>>(hs2, ln2w, xn2);

  // 7) fused dual GEMM: h = silu(x2@wg) * (x2@wu) -> bf16
  gemm_kernel<2><<<dim3(DFF / 128, M / 128), blk256, 0, stream>>>(
      xn2, wgate, wup, nullptr, nullptr, hgl, DFF, DMODEL);

  // 8) out = hs2 + h @ wdown
  gemm_kernel<1><<<dim3(DMODEL / 128, M / 128), blk256, 0, stream>>>(
      hgl, wdown, nullptr, hs2, out, nullptr, DMODEL, DFF);

  (void)n_in; (void)in_sizes; (void)out_size; (void)ws_size;
}
